// ProteinLMGraph_72318659330280
// MI455X (gfx1250) — matrix operand order repaired and checked by reading, not compiled, not run on hardware
//
#include <hip/hip_runtime.h>
#include <hip/hip_bf16.h>

typedef __attribute__((ext_vector_type(16))) _Float16 v16h;
typedef __attribute__((ext_vector_type(8)))  _Float16 v8h;
typedef __attribute__((ext_vector_type(8)))  float    v8f;

#define NEG_SLOPE 0.01f
#define LN_EPS 1e-5f

__device__ __forceinline__ float leakyf(float x) { return x >= 0.f ? x : NEG_SLOPE * x; }

// float atomic-max via signed/unsigned int trick (init must be -inf)
__device__ __forceinline__ void atomicMaxF(float* addr, float val) {
  if (val >= 0.f) atomicMax((int*)addr, __float_as_int(val));
  else            atomicMin((unsigned int*)addr, __float_as_uint(val));
}

// ---------------------------------------------------------------- utility
__global__ void k_copy2(const float* __restrict__ src, float* __restrict__ d1,
                        float* __restrict__ d2, int n) {
  int i = blockIdx.x * blockDim.x + threadIdx.x;
  if (i < n) { float v = src[i]; d1[i] = v; d2[i] = v; }
}

__global__ void k_zero(float* __restrict__ p, int n) {
  int i = blockIdx.x * blockDim.x + threadIdx.x;
  if (i < n) p[i] = 0.f;
}

__global__ void k_f32_to_f16(const float* __restrict__ src, _Float16* __restrict__ dst,
                             float scale, int n) {
  int i = blockIdx.x * blockDim.x + threadIdx.x;
  if (i < n) dst[i] = (_Float16)(src[i] * scale);
}

// w[K][N] (row-major) -> wt[N][K] as f16
__global__ void k_transpose_f16(const float* __restrict__ w, _Float16* __restrict__ wt,
                                int K, int N) {
  int i = blockIdx.x * blockDim.x + threadIdx.x;
  if (i >= K * N) return;
  int k = i / N, n = i % N;
  wt[(size_t)n * K + k] = (_Float16)w[i];
}

// ---------------------------------------------------------------- surface path
// stats[0:64]=sum, stats[64:128]=sumsq of t = surf_feats @ surf_up_w + b
__global__ void k_surf_stats(const float* __restrict__ sf, const float* __restrict__ w,
                             const float* __restrict__ b, float* __restrict__ stats, int n) {
  int t = threadIdx.x;
  int c = t & 63;
  int i = blockIdx.x * 4 + (t >> 6);
  float v = 0.f;
  bool ok = i < n;
  if (ok) {
    const float* x = sf + (size_t)i * 3;
    v = x[0] * w[c] + x[1] * w[64 + c] + x[2] * w[128 + c] + b[c];
  }
  __shared__ float r1[256], r2[256];
  r1[t] = ok ? v : 0.f;
  r2[t] = ok ? v * v : 0.f;
  __syncthreads();
  if (t < 128) { r1[t] += r1[t + 128]; r2[t] += r2[t + 128]; }
  __syncthreads();
  if (t < 64) {
    atomicAdd(&stats[c],      r1[t] + r1[t + 64]);
    atomicAdd(&stats[64 + c], r2[t] + r2[t + 64]);
  }
}

__global__ void k_surf_scatter(const float* __restrict__ sf, const float* __restrict__ w,
                               const float* __restrict__ b, const float* __restrict__ g,
                               const float* __restrict__ beta, const float* __restrict__ stats,
                               const int* __restrict__ surf_res, float* __restrict__ node_sum,
                               float* __restrict__ node_cnt, int n) {
  int tid = blockIdx.x * blockDim.x + threadIdx.x;
  int i = tid >> 6, c = tid & 63;
  if (i >= n) return;
  const float* x = sf + (size_t)i * 3;
  float v = x[0] * w[c] + x[1] * w[64 + c] + x[2] * w[128 + c] + b[c];
  float mean = stats[c] / (float)n;
  float var  = stats[64 + c] / (float)n - mean * mean;
  float val  = leakyf((v - mean) * rsqrtf(var + LN_EPS) * g[c] + beta[c]);
  int node = surf_res[i];
  atomicAdd(&node_sum[(size_t)node * 64 + c], val);
  if (c == 0) atomicAdd(&node_cnt[node], 1.f);
}

__global__ void k_gather_s(const float* __restrict__ node_sum, const float* __restrict__ node_cnt,
                           const int* __restrict__ uniq, float* __restrict__ s, int nu) {
  int tid = blockIdx.x * blockDim.x + threadIdx.x;
  int i = tid >> 6, c = tid & 63;
  if (i >= nu) return;
  int node = uniq[i];
  s[(size_t)i * 64 + c] = node_sum[(size_t)node * 64 + c] / fmaxf(node_cnt[node], 1.f);
}

// ---------------------------------------------------------------- GAT layer
__global__ void k_layer_init(float* __restrict__ agg, float* __restrict__ mx,
                             float* __restrict__ den, int nn, int dim) {
  int i = blockIdx.x * blockDim.x + threadIdx.x;
  if (i < nn * dim) agg[i] = 0.f;
  if (i < nn) { mx[i] = __int_as_float(0xFF800000u); den[i] = 0.f; }
}

// one wave per node: au = feat·awu, av = feat·awv
__global__ void k_alpha(const float* __restrict__ feat, const float* __restrict__ awu,
                        const float* __restrict__ awv, float* __restrict__ au,
                        float* __restrict__ av, int nn, int dim) {
  int wid = (blockIdx.x * blockDim.x + threadIdx.x) >> 5;
  int lane = threadIdx.x & 31;
  if (wid >= nn) return;
  const float* f = feat + (size_t)wid * dim;
  float du = 0.f, dv = 0.f;
  for (int j = lane; j < dim; j += 32) {
    float x = f[j];
    du += x * awu[j];
    dv += x * awv[j];
  }
  for (int o = 16; o; o >>= 1) {
    du += __shfl_xor(du, o, 32);
    dv += __shfl_xor(dv, o, 32);
  }
  if (lane == 0) { au[wid] = du; av[wid] = dv; }
}

__global__ void k_edge_max(const int* __restrict__ eu, const int* __restrict__ ev,
                           const float* __restrict__ au, const float* __restrict__ av,
                           float* __restrict__ elog, float* __restrict__ mx, int ne) {
  int e = blockIdx.x * blockDim.x + threadIdx.x;
  if (e >= ne) return;
  float lg = au[eu[e]] + av[ev[e]];
  elog[e] = lg;
  atomicMaxF(&mx[ev[e]], lg);
}

__global__ void k_edge_exp(const int* __restrict__ ev, float* __restrict__ elog,
                           const float* __restrict__ mx, float* __restrict__ den, int ne) {
  int e = blockIdx.x * blockDim.x + threadIdx.x;
  if (e >= ne) return;
  int v = ev[e];
  float ex = __expf(elog[e] - mx[v]);
  elog[e] = ex;
  atomicAdd(&den[v], ex);
}

// block per edge: agg[v] += feat[u] * w
__global__ void k_edge_agg(const int* __restrict__ eu, const int* __restrict__ ev,
                           const float* __restrict__ elog, const float* __restrict__ den,
                           const float* __restrict__ feat, float* __restrict__ agg, int dim) {
  int e = blockIdx.x;
  int u = eu[e], v = ev[e];
  float w = elog[e] / den[v];
  const float* fu = feat + (size_t)u * dim;
  float* av = agg + (size_t)v * dim;
  for (int d = threadIdx.x; d < dim; d += blockDim.x)
    atomicAdd(&av[d], fu[d] * w);
}

// ---------------------------------------------------------------- WMMA GEMM
// C[M x N] = A[M x K] (f16) @ Bt[N x K]^T (f16, pre-transposed) + bias[N]
// One wave computes a 32x32 output block = 2x2 WMMA tiles (4 x v8f accumulators).
// Register tiling: each A/B fragment is reused by 2 WMMAs -> 2 b128 loads per
// WMMA instead of 4, doubling FLOPs per VMEM issue.
// Per-lane A layout (ISA 7.12.2): row = lane&15; lanes<16 hold K {0..7,16..23},
// lanes>=16 hold K {8..15,24..31}. B: same K order as A, N = lane&15.
__global__ void k_wmma_gemm(const _Float16* __restrict__ A, const _Float16* __restrict__ Bt,
                            const float* __restrict__ bias, float* __restrict__ C,
                            int M, int N, int K) {
  int gw = (int)((blockIdx.x * blockDim.x + threadIdx.x) >> 5);
  int lane = threadIdx.x & 31;
  int tn = N >> 5;                       // 32-wide col blocks
  int tiles = (M >> 5) * tn;             // 32-tall row blocks
  if (gw >= tiles) return;               // whole-wave uniform exit: EXEC stays all-1s
  int m0 = (gw / tn) << 5;
  int n0 = (gw % tn) << 5;
  int half = lane >> 4;
  int l15 = lane & 15;
  const _Float16* a0p = A + (size_t)(m0 + l15) * (size_t)K;
  const _Float16* a1p = a0p + (size_t)16 * (size_t)K;
  const _Float16* b0p = Bt + (size_t)(n0 + l15) * (size_t)K;
  const _Float16* b1p = b0p + (size_t)16 * (size_t)K;
  v8f c00 = {}, c01 = {}, c10 = {}, c11 = {};
  int aoff = 8 * half;                   // A frag: halves at k0+aoff and k0+16+aoff
  int boff = 8 * half;                   // B frag: halves at k0+boff and k0+16+boff
  for (int k0 = 0; k0 < K; k0 += 32) {
    v16h a0, a1, b0, b1;
    *reinterpret_cast<v8h*>(&a0)       = *reinterpret_cast<const v8h*>(a0p + k0 + aoff);
    *(reinterpret_cast<v8h*>(&a0) + 1) = *reinterpret_cast<const v8h*>(a0p + k0 + 16 + aoff);
    *reinterpret_cast<v8h*>(&a1)       = *reinterpret_cast<const v8h*>(a1p + k0 + aoff);
    *(reinterpret_cast<v8h*>(&a1) + 1) = *reinterpret_cast<const v8h*>(a1p + k0 + 16 + aoff);
    *reinterpret_cast<v8h*>(&b0)       = *reinterpret_cast<const v8h*>(b0p + k0 + boff);
    *(reinterpret_cast<v8h*>(&b0) + 1) = *reinterpret_cast<const v8h*>(b0p + k0 + boff + 16);
    *reinterpret_cast<v8h*>(&b1)       = *reinterpret_cast<const v8h*>(b1p + k0 + boff);
    *(reinterpret_cast<v8h*>(&b1) + 1) = *reinterpret_cast<const v8h*>(b1p + k0 + boff + 16);
    c00 = __builtin_amdgcn_wmma_f32_16x16x32_f16(false, a0, false, b0, (short)0, c00, false, false);
    c01 = __builtin_amdgcn_wmma_f32_16x16x32_f16(false, a0, false, b1, (short)0, c01, false, false);
    c10 = __builtin_amdgcn_wmma_f32_16x16x32_f16(false, a1, false, b0, (short)0, c10, false, false);
    c11 = __builtin_amdgcn_wmma_f32_16x16x32_f16(false, a1, false, b1, (short)0, c11, false, false);
  }
  int colL = n0 + l15;
  int colH = n0 + 16 + l15;
  float bvL = bias ? bias[colL] : 0.f;
  float bvH = bias ? bias[colH] : 0.f;
  float* r0 = C + (size_t)(m0 + 8 * half) * (size_t)N;       // rows m0..  (tiles 00/01)
  float* r1 = C + (size_t)(m0 + 16 + 8 * half) * (size_t)N;  // rows m0+16 (tiles 10/11)
  for (int r = 0; r < 8; ++r) {
    r0[(size_t)r * N + colL] = c00[r] + bvL;
    r0[(size_t)r * N + colH] = c01[r] + bvH;
    r1[(size_t)r * N + colL] = c10[r] + bvL;
    r1[(size_t)r * N + colH] = c11[r] + bvH;
  }
}

// ---------------------------------------------------------------- layer norm + residual
__global__ void k_ln_acc(const float* __restrict__ x, const float* __restrict__ g,
                         const float* __restrict__ b, float* __restrict__ feat,
                         float* __restrict__ acc, int dim) {
  int row = blockIdx.x;
  const float* xr = x + (size_t)row * dim;
  __shared__ float sd[256];
  float s = 0.f, s2 = 0.f;
  for (int j = threadIdx.x; j < dim; j += 256) { float v = xr[j]; s += v; s2 += v * v; }
  sd[threadIdx.x] = s; __syncthreads();
  for (int o = 128; o > 0; o >>= 1) { if (threadIdx.x < o) sd[threadIdx.x] += sd[threadIdx.x + o]; __syncthreads(); }
  float sum = sd[0]; __syncthreads();
  sd[threadIdx.x] = s2; __syncthreads();
  for (int o = 128; o > 0; o >>= 1) { if (threadIdx.x < o) sd[threadIdx.x] += sd[threadIdx.x + o]; __syncthreads(); }
  float sum2 = sd[0];
  float mu = sum / (float)dim;
  float var = sum2 / (float)dim - mu * mu;
  float inv = rsqrtf(var + LN_EPS);
  for (int j = threadIdx.x; j < dim; j += 256) {
    float v = leakyf((xr[j] - mu) * inv * g[j] + b[j]);
    feat[(size_t)row * dim + j] = v;
    acc[(size_t)row * dim + j] += v;
  }
}

// ---------------------------------------------------------------- graph pooling + LN -> prot2
__global__ void k_prot(const float* __restrict__ nf, const float* __restrict__ g,
                       const float* __restrict__ b, float* __restrict__ out, int rpg) {
  int gph = blockIdx.x;
  int t = threadIdx.x;
  int c = t & 63, seg = t >> 6;
  float s = 0.f;
  for (int r = seg; r < rpg; r += 4)
    s += nf[((size_t)(gph * rpg + r)) * 64 + c];
  __shared__ float red[256];
  red[t] = s; __syncthreads();
  if (t < 64) red[t] = (red[t] + red[t + 64] + red[t + 128] + red[t + 192]) / (float)rpg;
  __syncthreads();
  __shared__ float mu_s, inv_s;
  if (t == 0) {
    float ss = 0.f, ss2 = 0.f;
    for (int j = 0; j < 64; ++j) { float p = red[j]; ss += p; ss2 += p * p; }
    float mu = ss / 64.f;
    mu_s = mu;
    inv_s = rsqrtf(ss2 / 64.f - mu * mu + LN_EPS);
  }
  __syncthreads();
  if (t < 64)
    out[gph * 64 + t] = leakyf((red[t] - mu_s) * inv_s * g[t] + b[t]);
}

// ---------------------------------------------------------------- down path
// y[i,c] = concat(nf[uniq[i]], s[i]) . down_w[:,c] + b[c]; stats at [128..255]
__global__ void k_down_mm(const float* __restrict__ nf, const float* __restrict__ s,
                          const int* __restrict__ uniq, const float* __restrict__ dw,
                          const float* __restrict__ db, float* __restrict__ y,
                          float* __restrict__ stats, int nu) {
  int t = threadIdx.x;
  int c = t & 63;
  int i = blockIdx.x * 4 + (t >> 6);
  float d = 0.f;
  bool ok = i < nu;
  if (ok) {
    d = db[c];
    const float* nfr = nf + (size_t)uniq[i] * 64;
    const float* sr = s + (size_t)i * 64;
    for (int j = 0; j < 64; ++j) d += nfr[j] * dw[j * 64 + c];
    for (int j = 0; j < 64; ++j) d += sr[j] * dw[(64 + j) * 64 + c];
    y[(size_t)i * 64 + c] = d;
  }
  __shared__ float r1[256], r2[256];
  r1[t] = ok ? d : 0.f;
  r2[t] = ok ? d * d : 0.f;
  __syncthreads();
  if (t < 128) { r1[t] += r1[t + 128]; r2[t] += r2[t + 128]; }
  __syncthreads();
  if (t < 64) {
    atomicAdd(&stats[128 + c], r1[t] + r1[t + 64]);
    atomicAdd(&stats[192 + c], r2[t] + r2[t + 64]);
  }
}

__global__ void k_down_scatter(const float* __restrict__ y, const float* __restrict__ stats,
                               const float* __restrict__ g, const float* __restrict__ beta,
                               const int* __restrict__ rb, float* __restrict__ gsum,
                               float* __restrict__ gcnt, int nu) {
  int tid = blockIdx.x * blockDim.x + threadIdx.x;
  int i = tid >> 6, c = tid & 63;
  if (i >= nu) return;
  float mean = stats[128 + c] / (float)nu;
  float var  = stats[192 + c] / (float)nu - mean * mean;
  float v = leakyf((y[(size_t)i * 64 + c] - mean) * rsqrtf(var + LN_EPS) * g[c] + beta[c]);
  int gph = rb[i];
  atomicAdd(&gsum[gph * 64 + c], v);
  if (c == 0) atomicAdd(&gcnt[gph], 1.f);
}

// one block of 64 threads: per-channel BN over the 16 graphs -> ns2
__global__ void k_ns2(const float* __restrict__ gsum, const float* __restrict__ gcnt,
                      const float* __restrict__ g, const float* __restrict__ b,
                      float* __restrict__ out) {
  int c = threadIdx.x;
  if (c >= 64) return;
  float vals[16];
  float s = 0.f, s2 = 0.f;
  for (int gr = 0; gr < 16; ++gr) {
    float v = gsum[gr * 64 + c] / fmaxf(gcnt[gr], 1.f);
    vals[gr] = v; s += v; s2 += v * v;
  }
  float mu = s / 16.f;
  float inv = rsqrtf(s2 / 16.f - mu * mu + LN_EPS);
  for (int gr = 0; gr < 16; ++gr)
    out[1024 + gr * 64 + c] = leakyf((vals[gr] - mu) * inv * g[c] + b[c]);
}

// ---------------------------------------------------------------- host
extern "C" void kernel_launch(void* const* d_in, const int* in_sizes, int n_in,
                              void* d_out, int out_size, void* d_ws, size_t ws_size,
                              hipStream_t stream) {
  const int N = 8192, E = 65536, DIM = 1024, HID = 64, NS = 40000, NU = 6000, G = 16,
            RPG = 512, NL = 4;

  const int*   edge_index = (const int*)d_in[0];
  const int*   eu = edge_index;
  const int*   ev = edge_index + E;
  const float* n_feats     = (const float*)d_in[1];
  const float* surf_feats  = (const float*)d_in[3];
  const int*   surf_res    = (const int*)d_in[4];
  const int*   res_batch   = (const int*)d_in[5];
  const int*   surf_set    = (const int*)d_in[6];
  const float* surf_up_w   = (const float*)d_in[7];
  const float* surf_up_b   = (const float*)d_in[8];
  const float* surf_up_g   = (const float*)d_in[9];
  const float* surf_up_be  = (const float*)d_in[10];
  const float* a_w         = (const float*)d_in[11];
  const float* lin_w       = (const float*)d_in[12];
  const float* lin_b       = (const float*)d_in[13];
  const float* ln_g        = (const float*)d_in[14];
  const float* ln_b        = (const float*)d_in[15];
  const float* hid_w       = (const float*)d_in[16];
  const float* hid_b       = (const float*)d_in[17];
  const float* down_w      = (const float*)d_in[18];
  const float* down_b      = (const float*)d_in[19];
  const float* down_g      = (const float*)d_in[20];
  const float* down_be     = (const float*)d_in[21];
  const float* prot_ln_g   = (const float*)d_in[22];
  const float* prot_ln_b   = (const float*)d_in[23];
  const float* surf_bn_g   = (const float*)d_in[24];
  const float* surf_bn_b   = (const float*)d_in[25];
  float* out = (float*)d_out;

  // ---- workspace bump allocator (~128 MB total)
  char* wsp = (char*)d_ws;
  auto alloc = [&](size_t bytes) -> void* {
    void* p = (void*)wsp;
    wsp += (bytes + 255) & ~(size_t)255;
    return p;
  };
  float*     feat     = (float*)alloc((size_t)N * DIM * 4);
  float*     acc      = (float*)alloc((size_t)N * DIM * 4);
  float*     agg      = (float*)alloc((size_t)N * DIM * 4);  // also GEMM output (reads only f16 copy)
  _Float16*  ah       = (_Float16*)alloc((size_t)N * DIM * 2);
  _Float16*  wt       = (_Float16*)alloc((size_t)DIM * DIM * 2);
  float*     nf       = (float*)alloc((size_t)N * HID * 4);
  // accumulators zeroed as one contiguous span:
  float*     node_sum = (float*)alloc((size_t)N * HID * 4);
  float*     node_cnt = (float*)alloc((size_t)N * 4);
  float*     stats    = (float*)alloc(256 * 4);
  float*     gsum     = (float*)alloc((size_t)G * HID * 4);
  float*     gcnt     = (float*)alloc((size_t)G * 4);
  char*      zero_end = wsp;
  float*     au       = (float*)alloc((size_t)N * 4);
  float*     av       = (float*)alloc((size_t)N * 4);
  float*     mx       = (float*)alloc((size_t)N * 4);
  float*     den      = (float*)alloc((size_t)N * 4);
  float*     elog     = (float*)alloc((size_t)E * 4);
  float*     sbuf     = (float*)alloc((size_t)NU * HID * 4);
  float*     ybuf     = (float*)alloc((size_t)NU * HID * 4);

  // init feat/acc; zero the atomic accumulator span
  k_copy2<<<(N * DIM + 255) / 256, 256, 0, stream>>>(n_feats, feat, acc, N * DIM);
  int zn = (int)((zero_end - (char*)node_sum) / 4);
  k_zero<<<(zn + 255) / 256, 256, 0, stream>>>(node_sum, zn);

  // ---- surface path
  k_surf_stats<<<NS / 4, 256, 0, stream>>>(surf_feats, surf_up_w, surf_up_b, stats, NS);
  k_surf_scatter<<<(NS * HID + 255) / 256, 256, 0, stream>>>(
      surf_feats, surf_up_w, surf_up_b, surf_up_g, surf_up_be, stats, surf_res,
      node_sum, node_cnt, NS);
  k_gather_s<<<(NU * HID + 255) / 256, 256, 0, stream>>>(node_sum, node_cnt, surf_set, sbuf, NU);

  // ---- GAT layers
  for (int l = 0; l < NL; ++l) {
    k_layer_init<<<(N * DIM + 255) / 256, 256, 0, stream>>>(agg, mx, den, N, DIM);
    k_alpha<<<N / 8, 256, 0, stream>>>(feat, a_w + (size_t)l * 2 * DIM,
                                       a_w + (size_t)l * 2 * DIM + DIM, au, av, N, DIM);
    k_edge_max<<<E / 256, 256, 0, stream>>>(eu, ev, au, av, elog, mx, E);
    k_edge_exp<<<E / 256, 256, 0, stream>>>(ev, elog, mx, den, E);
    k_edge_agg<<<E, 256, 0, stream>>>(eu, ev, elog, den, feat, agg, DIM);
    k_f32_to_f16<<<(N * DIM + 255) / 256, 256, 0, stream>>>(agg, ah, 1.f, N * DIM);
    k_transpose_f16<<<(DIM * DIM + 255) / 256, 256, 0, stream>>>(
        lin_w + (size_t)l * DIM * DIM, wt, DIM, DIM);
    int waves = (N / 32) * (DIM / 32);   // 32x32 block per wave
    k_wmma_gemm<<<(waves * 32 + 255) / 256, 256, 0, stream>>>(
        ah, wt, lin_b + (size_t)l * DIM, agg, N, DIM, DIM);
    k_ln_acc<<<N, 256, 0, stream>>>(agg, ln_g + (size_t)l * DIM, ln_b + (size_t)l * DIM,
                                    feat, acc, DIM);
  }

  // ---- nf = (acc/5) @ hid_w + hid_b  (WMMA, scale folded into f16 convert)
  k_f32_to_f16<<<(N * DIM + 255) / 256, 256, 0, stream>>>(acc, ah, 1.f / (NL + 1), N * DIM);
  k_transpose_f16<<<(DIM * HID + 255) / 256, 256, 0, stream>>>(hid_w, wt, DIM, HID);
  {
    int waves = (N / 32) * (HID / 32);   // 32x32 block per wave
    k_wmma_gemm<<<(waves * 32 + 255) / 256, 256, 0, stream>>>(ah, wt, hid_b, nf, N, HID, DIM);
  }

  // ---- prot2 (out[0:1024])
  k_prot<<<G, 256, 0, stream>>>(nf, prot_ln_g, prot_ln_b, out, RPG);

  // ---- ns2 (out[1024:2048])
  k_down_mm<<<NU / 4, 256, 0, stream>>>(nf, sbuf, surf_set, down_w, down_b, ybuf, stats, NU);
  k_down_scatter<<<(NU * HID + 255) / 256, 256, 0, stream>>>(
      ybuf, stats, down_g, down_be, res_batch, gsum, gcnt, NU);
  k_ns2<<<1, 64, 0, stream>>>(gsum, gcnt, surf_bn_g, surf_bn_b, out);
}
